// DensityDenoiser_30580167147702
// MI455X (gfx1250) — matrix operand order repaired and checked by reading, not compiled, not run on hardware
//
#include <hip/hip_runtime.h>
#include <hip/hip_bf16.h>

typedef __attribute__((ext_vector_type(16))) _Float16 v16h;
typedef __attribute__((ext_vector_type(8)))  _Float16 v8h;
typedef __attribute__((ext_vector_type(8)))  float    v8f;

#define NN        50000
#define NE        800000
#define NINV      96
#define HDIM      128
#define IN_DIM    320
#define NTILES    (NE / 16)          // 50000
#define NPAIRS    (NTILES / 2)       // 25000 (each wave does 32 edges / iter)
#define NTHREADS  256
#define WPW       8                  // waves per workgroup (wave32)

#define W1_HALFS  (IN_DIM * HDIM)    // 40960 f16
#define W2_HALFS  (HDIM * HDIM)      // 16384 f16
#define H1_STRIDE 136                // 128 + 8 pad halfwords -> conflict-free b128 reads
#define H1_WAVE   (16 * H1_STRIDE)   // per-wave h1 scratch (halfwords), reused for both tiles

__global__ __launch_bounds__(NTHREADS) void edge_mlp_ln_kernel(
    const float* __restrict__ NI,    // [NN, 96]
    const float* __restrict__ EF,    // [NE, 128]
    const int*   __restrict__ EIDX,  // [2, NE]
    const float* __restrict__ W1,    // [320, 128]
    const float* __restrict__ B1,    // [128]
    const float* __restrict__ W2,    // [128, 128]
    const float* __restrict__ B2,    // [128]
    const float* __restrict__ GAMMA, // [128]
    const float* __restrict__ BETA,  // [128]
    float* __restrict__ OUT)         // [NE, 128]
{
    // LDS: packed W1 fragments | packed W2 fragments | per-wave h1 scratch
    __shared__ __align__(16) _Float16 lds[W1_HALFS + W2_HALFS + WPW * H1_WAVE];

    const int tid = threadIdx.x;

    // ---- Phase 1: pack weights (f32 -> f16) into WMMA B-fragment order ----
    // p = ((kstep*8 + ntile)*32 + lane)*16 + j ; lane L holds B[k][n] with
    // n = ntile*16 + (L&15), k = kstep*32 + (j>>3)*16 + (L>>4)*8 + (j&7)
    for (int p = tid; p < W1_HALFS; p += NTHREADS) {
        int j    = p & 15;
        int lane = (p >> 4) & 31;
        int nt   = (p >> 9) & 7;
        int ks   = p >> 12;                      // 0..9
        int k    = ks * 32 + (j >> 3) * 16 + (lane >> 4) * 8 + (j & 7);
        int n    = nt * 16 + (lane & 15);
        lds[p] = (_Float16)W1[k * HDIM + n];
    }
    for (int p = tid; p < W2_HALFS; p += NTHREADS) {
        int j    = p & 15;
        int lane = (p >> 4) & 31;
        int nt   = (p >> 9) & 7;
        int ks   = p >> 12;                      // 0..3
        int k    = ks * 32 + (j >> 3) * 16 + (lane >> 4) * 8 + (j & 7);
        int n    = nt * 16 + (lane & 15);
        lds[W1_HALFS + p] = (_Float16)W2[k * HDIM + n];
    }
    __syncthreads();

    const int lane = tid & 31;
    const int wid  = tid >> 5;
    const int hi   = lane >> 4;    // 16-lane half
    const int lo   = lane & 15;    // M row for A frags / N column for B,C,D frags
    _Float16* h1 = &lds[W1_HALFS + W2_HALFS + wid * H1_WAVE];

    // Per-lane invariant params: n = nt*16 + lo
    float b1v[8], b2v[8], gv[8], btv[8];
    #pragma unroll
    for (int nt = 0; nt < 8; ++nt) {
        int n = nt * 16 + lo;
        b1v[nt] = B1[n]; b2v[nt] = B2[n]; gv[nt] = GAMMA[n]; btv[nt] = BETA[n];
    }

    const int totWaves = gridDim.x * WPW;
    for (int pr = blockIdx.x * WPW + wid; pr < NPAIRS; pr += totWaves) {
        const int e0 = pr * 32;                    // 32 edges per iteration
        // Per-lane edge rows for the two 16-edge M-tiles
        const int ea = e0 + lo;
        const int eb = e0 + 16 + lo;
        const int sia = EIDX[ea],      dia = EIDX[NE + ea];
        const int sib = EIDX[eb],      dib = EIDX[NE + eb];
        const float* srcpa = NI + (size_t)sia * NINV;
        const float* dstpa = NI + (size_t)dia * NINV;
        const float* efpa  = EF + (size_t)ea  * HDIM;
        const float* srcpb = NI + (size_t)sib * NINV;
        const float* dstpb = NI + (size_t)dib * NINV;
        const float* efpb  = EF + (size_t)eb  * HDIM;

        // ------------- Layer 1: two 16-row A tiles share every B frag -------
        v8f acc[2][8] = {};
        #pragma unroll
        for (int ks = 0; ks < 10; ++ks) {
            // concat boundaries (96,192) are multiples of 32 -> uniform source
            const float *base0, *base1;
            if      (ks < 3) { base0 = srcpa + ks * 32;       base1 = srcpb + ks * 32; }
            else if (ks < 6) { base0 = dstpa + (ks - 3) * 32; base1 = dstpb + (ks - 3) * 32; }
            else             { base0 = efpa  + (ks - 6) * 32; base1 = efpb  + (ks - 6) * 32; }
            v16h a0, a1;
            {
                const float* c0 = base0 + 8 * hi;
                const float* c1 = base0 + 16 + 8 * hi;
                #pragma unroll
                for (int j = 0; j < 8; ++j) { a0[j] = (_Float16)c0[j]; a0[8 + j] = (_Float16)c1[j]; }
            }
            {
                const float* c0 = base1 + 8 * hi;
                const float* c1 = base1 + 16 + 8 * hi;
                #pragma unroll
                for (int j = 0; j < 8; ++j) { a1[j] = (_Float16)c0[j]; a1[8 + j] = (_Float16)c1[j]; }
            }
            #pragma unroll
            for (int nt = 0; nt < 8; ++nt) {
                const _Float16* bp = &lds[((ks * 8 + nt) * 32 + lane) * 16];
                v16h b;
                *(v8h*)&b         = *(const v8h*)bp;
                *(((v8h*)&b) + 1) = *(const v8h*)(bp + 8);
                acc[0][nt] = __builtin_amdgcn_wmma_f32_16x16x32_f16(
                    false, a0, false, b, (short)0, acc[0][nt], false, false);
                acc[1][nt] = __builtin_amdgcn_wmma_f32_16x16x32_f16(
                    false, a1, false, b, (short)0, acc[1][nt], false, false);
            }
        }

        // ------------- Per 16-edge tile: layer 2 + LayerNorm + store --------
        #pragma unroll
        for (int tt = 0; tt < 2; ++tt) {
            const int et = e0 + tt * 16;

            // bias + ReLU, write h1 to LDS (C-layout -> row-major f16)
            #pragma unroll
            for (int nt = 0; nt < 8; ++nt) {
                #pragma unroll
                for (int v = 0; v < 8; ++v) {
                    float x = acc[tt][nt][v] + b1v[nt];
                    x = x > 0.f ? x : 0.f;
                    int m = v + 8 * hi;
                    h1[m * H1_STRIDE + nt * 16 + lo] = (_Float16)x;
                }
            }
            asm volatile("s_wait_dscnt 0" ::: "memory"); // stores visible to our loads

            v8f acc2[8] = {};
            #pragma unroll
            for (int ks = 0; ks < 4; ++ks) {
                const _Float16* r = &h1[lo * H1_STRIDE + ks * 32 + 8 * hi];
                v16h a;
                *(v8h*)&a         = *(const v8h*)r;
                *(((v8h*)&a) + 1) = *(const v8h*)(r + 16);
                #pragma unroll
                for (int nt = 0; nt < 8; ++nt) {
                    const _Float16* bp = &lds[W1_HALFS + ((ks * 8 + nt) * 32 + lane) * 16];
                    v16h b;
                    *(v8h*)&b         = *(const v8h*)bp;
                    *(((v8h*)&b) + 1) = *(const v8h*)(bp + 8);
                    acc2[nt] = __builtin_amdgcn_wmma_f32_16x16x32_f16(
                        false, a, false, b, (short)0, acc2[nt], false, false);
                }
            }
            asm volatile("" ::: "memory");   // WAR fence vs next tile's h1 stores

            // bias2 + LayerNorm + residual + store
            #pragma unroll
            for (int v = 0; v < 8; ++v) {
                float xv[8];
                float s = 0.f, s2 = 0.f;
                #pragma unroll
                for (int nt = 0; nt < 8; ++nt) {
                    float x = acc2[nt][v] + b2v[nt];
                    xv[nt] = x; s += x; s2 += x * x;
                }
                #pragma unroll
                for (int off = 1; off < 16; off <<= 1) {
                    s  += __shfl_xor(s,  off, 32);
                    s2 += __shfl_xor(s2, off, 32);
                }
                float mean = s * (1.f / 128.f);
                float var  = s2 * (1.f / 128.f) - mean * mean;
                float inv  = rsqrtf(var + 1e-5f);
                int m = v + 8 * hi;
                const float* efr = EF  + (size_t)(et + m) * HDIM;
                float*      outr = OUT + (size_t)(et + m) * HDIM;
                #pragma unroll
                for (int nt = 0; nt < 8; ++nt) {
                    int n = nt * 16 + lo;
                    outr[n] = efr[n] + (xv[nt] - mean) * inv * gv[nt] + btv[nt];
                }
            }
        }
    }
}

extern "C" void kernel_launch(void* const* d_in, const int* in_sizes, int n_in,
                              void* d_out, int out_size, void* d_ws, size_t ws_size,
                              hipStream_t stream) {
    const float* NI    = (const float*)d_in[0];
    const float* EF    = (const float*)d_in[1];
    const int*   EIDX  = (const int*)  d_in[2];
    const float* W1    = (const float*)d_in[3];
    const float* B1    = (const float*)d_in[4];
    const float* W2    = (const float*)d_in[5];
    const float* B2    = (const float*)d_in[6];
    const float* GAMMA = (const float*)d_in[7];
    const float* BETA  = (const float*)d_in[8];
    float* OUT = (float*)d_out;

    (void)in_sizes; (void)n_in; (void)out_size; (void)d_ws; (void)ws_size;

    dim3 grid(1024), block(NTHREADS);
    edge_mlp_ln_kernel<<<grid, block, 0, stream>>>(
        NI, EF, EIDX, W1, B1, W2, B2, GAMMA, BETA, OUT);
}
